// MultiHeadedAttention_5978594476345
// MI455X (gfx1250) — matrix operand order repaired and checked by reading, not compiled, not run on hardware
//
#include <hip/hip_runtime.h>
#include <hip/hip_bf16.h>

typedef __bf16 bf16_t;
typedef __attribute__((ext_vector_type(16))) __bf16 v16bf;
typedef __attribute__((ext_vector_type(8)))  __bf16 v8bf;
typedef __attribute__((ext_vector_type(8)))  float  v8f;
typedef __attribute__((ext_vector_type(4)))  int    v4i;

#define DEV static __device__ __forceinline__
#define AS1 __attribute__((address_space(1)))
#define AS3 __attribute__((address_space(3)))

constexpr int Bsz   = 4;
constexpr int Sq    = 2048;
constexpr int Dm    = 1024;
constexpr int Hh    = 16;
constexpr int Dk    = 64;
constexpr int Mrows = Bsz * Sq;          // 8192
constexpr float LOG2E = 1.44269504088896f;

#if __has_builtin(__builtin_amdgcn_global_load_async_to_lds_b128)
#define HAVE_ASYNC_LDS 1
#else
#define HAVE_ASYNC_LDS 0
#endif

// 16-byte global -> LDS copy: async (ASYNCcnt path) when available.
DEV void copy16_g2l(const bf16_t* g, bf16_t* l) {
#if HAVE_ASYNC_LDS
    __builtin_amdgcn_global_load_async_to_lds_b128(
        (AS1 v4i*)g, (AS3 v4i*)l, /*offset=*/0, /*cpol=*/0);
#else
    *(v8bf*)l = *(const v8bf*)g;
#endif
}

DEV void async_stage_wait() {
#if HAVE_ASYNC_LDS
#if __has_builtin(__builtin_amdgcn_s_wait_asynccnt)
    __builtin_amdgcn_s_wait_asynccnt(0);
#else
    asm volatile("s_wait_asynccnt 0x0" ::: "memory");
#endif
#endif
}

// Build a 16-element bf16 fragment from two 16-byte loads.
DEV v16bf ld_frag(const bf16_t* lo, const bf16_t* hi) {
    union { v16bf v; v8bf h[2]; } u;
    u.h[0] = *(const v8bf*)lo;
    u.h[1] = *(const v8bf*)hi;
    return u.v;
}

DEV v8f wmma_bf16(v16bf a, v16bf b, v8f c) {
    // D = A(16x32) * B(32x16) + C, f32 accumulate
    return __builtin_amdgcn_wmma_f32_16x16x32_bf16(
        /*neg_a=*/false, a, /*neg_b=*/false, b,
        /*c_mod=*/(short)0, c, /*reuse_a=*/false, /*reuse_b=*/false);
}

// ---------------------------------------------------------------------------
// Kernel 1: f32 -> bf16 weight pack (W is [N,K] row-major; kept as-is since
// that is exactly the WMMA B-operand feeding layout for y = x @ W^T).
// ---------------------------------------------------------------------------
__global__ __launch_bounds__(256)
void pack_bf16_kernel(const float* __restrict__ src, bf16_t* __restrict__ dst, int n4) {
    int i = blockIdx.x * 256 + threadIdx.x;
    if (i < n4) {
        float4 f = ((const float4*)src)[i];
        union { bf16_t b[4]; unsigned long long u; } o;
        o.b[0] = (bf16_t)f.x; o.b[1] = (bf16_t)f.y;
        o.b[2] = (bf16_t)f.z; o.b[3] = (bf16_t)f.w;
        ((unsigned long long*)dst)[i] = o.u;
    }
}

// ---------------------------------------------------------------------------
// Kernel 2/4: C[M,1024] = A[M,1024] @ W^T, W bf16 [N,K] row-major.
//   A_F32:       A is f32 (converted to bf16 while staging to LDS)
//   SPLIT_HEADS: write bf16 into [B,H,S,64] layout; else f32 row-major out.
// Block: 256 threads = 8 waves; tile 128x128, K-step 32.
// Wave grid 2(M) x 4(N); each wave owns 64x32 = 4x2 WMMA accumulators.
// ---------------------------------------------------------------------------
template <bool A_F32, bool SPLIT_HEADS>
__global__ __launch_bounds__(256)
void gemm_xw_kernel(const void* __restrict__ Aptr,
                    const bf16_t* __restrict__ Wb,
                    void* __restrict__ Outp) {
    constexpr int BM = 128, BK = 32, LSTR = 40; // 40 elems = 80B, 16B-multiple
    __shared__ __align__(16) bf16_t As[BM * LSTR];
    __shared__ __align__(16) bf16_t Ws[BM * LSTR];

    const int m0   = blockIdx.y * BM;
    const int n0   = blockIdx.x * BM;
    const int tid  = threadIdx.x;
    const int wid  = tid >> 5;
    const int ln   = tid & 15;
    const int half = (tid >> 4) & 1;
    const int waveM = wid & 1;   // 0..1  -> 64-row half
    const int waveN = wid >> 1;  // 0..3  -> 32-col strip

    v8f acc[4][2] = {};

    const int srow = tid >> 1;          // 0..127
    const int scol = (tid & 1) * 16;    // 0 or 16

    for (int k0 = 0; k0 < Dm; k0 += BK) {
        __syncthreads();
        // ---- stage W tile (128 N-rows x 32 K) : async global->LDS ----
        {
            const bf16_t* p = Wb + (size_t)(n0 + srow) * Dm + k0 + scol;
            copy16_g2l(p,     &Ws[srow * LSTR + scol]);
            copy16_g2l(p + 8, &Ws[srow * LSTR + scol + 8]);
        }
        // ---- stage A tile (128 x 32 bf16) ----
        if constexpr (A_F32) {
            const float* Af = (const float*)Aptr;
            const float4* p = (const float4*)(Af + (size_t)(m0 + srow) * Dm + k0 + scol);
            float4 f0 = p[0], f1 = p[1], f2 = p[2], f3 = p[3];
            bf16_t* d = &As[srow * LSTR + scol];
            d[0]  = (bf16_t)f0.x; d[1]  = (bf16_t)f0.y; d[2]  = (bf16_t)f0.z; d[3]  = (bf16_t)f0.w;
            d[4]  = (bf16_t)f1.x; d[5]  = (bf16_t)f1.y; d[6]  = (bf16_t)f1.z; d[7]  = (bf16_t)f1.w;
            d[8]  = (bf16_t)f2.x; d[9]  = (bf16_t)f2.y; d[10] = (bf16_t)f2.z; d[11] = (bf16_t)f2.w;
            d[12] = (bf16_t)f3.x; d[13] = (bf16_t)f3.y; d[14] = (bf16_t)f3.z; d[15] = (bf16_t)f3.w;
        } else {
            const bf16_t* Ab = (const bf16_t*)Aptr;
            const bf16_t* p = Ab + (size_t)(m0 + srow) * Dm + k0 + scol;
            copy16_g2l(p,     &As[srow * LSTR + scol]);
            copy16_g2l(p + 8, &As[srow * LSTR + scol + 8]);
        }
        async_stage_wait();
        __syncthreads();

        // ---- fragments + 8 WMMAs ----
        v16bf af[4], bw[2];
        #pragma unroll
        for (int mt = 0; mt < 4; ++mt) {
            const bf16_t* r = &As[(waveM * 64 + mt * 16 + ln) * LSTR];
            af[mt] = ld_frag(r + half * 8, r + 16 + half * 8);
        }
        #pragma unroll
        for (int nt = 0; nt < 2; ++nt) {
            const bf16_t* r = &Ws[(waveN * 32 + nt * 16 + ln) * LSTR];
            bw[nt] = ld_frag(r + half * 8, r + 16 + half * 8);
        }
        #pragma unroll
        for (int mt = 0; mt < 4; ++mt)
            #pragma unroll
            for (int nt = 0; nt < 2; ++nt)
                acc[mt][nt] = wmma_bf16(af[mt], bw[nt], acc[mt][nt]);
    }

    // ---- epilogue ----
    #pragma unroll
    for (int mt = 0; mt < 4; ++mt) {
        #pragma unroll
        for (int nt = 0; nt < 2; ++nt) {
            #pragma unroll
            for (int i = 0; i < 8; ++i) {
                int row = m0 + waveM * 64 + mt * 16 + i + half * 8;
                int col = n0 + waveN * 32 + nt * 16 + ln;
                float v = acc[mt][nt][i];
                if constexpr (SPLIT_HEADS) {
                    bf16_t* O = (bf16_t*)Outp;
                    int b = row >> 11, s = row & (Sq - 1);
                    int h = col >> 6,  d = col & (Dk - 1);
                    O[(((size_t)(b * Hh + h) * Sq) + s) * Dk + d] = (bf16_t)v;
                } else {
                    float* O = (float*)Outp;
                    O[(size_t)row * Dm + col] = v;
                }
            }
        }
    }
}

// ---------------------------------------------------------------------------
// Kernel 3: flash attention per (b, h, 128-query tile). 8 waves x 16 q-rows.
// Key chunks of 64; online softmax; bf16 WMMA for QK^T and PV.
// ---------------------------------------------------------------------------
__global__ __launch_bounds__(256)
void mha_flash_kernel(const bf16_t* __restrict__ Q, const bf16_t* __restrict__ K,
                      const bf16_t* __restrict__ V, const int* __restrict__ mask,
                      bf16_t* __restrict__ AO) {
    constexpr int QT = 128, KC = 64, LSTR = 72;  // 72 elems = 144B, 16B-multiple
    __shared__ __align__(16) bf16_t Ks [KC * LSTR];  // [key][d]   (B-operand for QK^T)
    __shared__ __align__(16) bf16_t Vts[Dk * LSTR];  // [d][key]   (B-operand for PV)
    __shared__ __align__(16) bf16_t Ps [8 * 16 * LSTR]; // per-wave P: [m][key]

    const int b = blockIdx.z, h = blockIdx.y;
    const int q0 = blockIdx.x * QT;
    const int tid = threadIdx.x, wid = tid >> 5;
    const int ln = tid & 15, half = (tid >> 4) & 1;
    const size_t bh = (size_t)(b * Hh + h);
    const bf16_t* Qb = Q + bh * Sq * Dk;
    const bf16_t* Kb = K + bh * Sq * Dk;
    const bf16_t* Vb = V + bh * Sq * Dk;

    // Q A-fragments for this wave's 16 rows (k = 0..63 -> two 16x32 frags)
    const bf16_t* qr = Qb + (size_t)(q0 + wid * 16 + ln) * Dk;
    v16bf qa[2];
    qa[0] = ld_frag(qr + half * 8,      qr + 16 + half * 8);
    qa[1] = ld_frag(qr + 32 + half * 8, qr + 48 + half * 8);

    v8f o[4] = {};
    float m_run[8], l_run[8];
    #pragma unroll
    for (int i = 0; i < 8; ++i) { m_run[i] = -3.0e38f; l_run[i] = 0.0f; }

    const int crow = tid >> 2;         // 0..63
    const int cseg = (tid & 3) * 16;   // 0,16,32,48
    bf16_t* Pw = &Ps[wid * 16 * LSTR];

    for (int kb = 0; kb < Sq; kb += KC) {
        __syncthreads();
        // ---- stage K chunk (async direct copy) and V chunk (transposed) ----
        {
            const bf16_t* kp = Kb + (size_t)(kb + crow) * Dk + cseg;
            copy16_g2l(kp,     &Ks[crow * LSTR + cseg]);
            copy16_g2l(kp + 8, &Ks[crow * LSTR + cseg + 8]);
            const v8bf* vp = (const v8bf*)(Vb + (size_t)(kb + crow) * Dk + cseg);
            v8bf v0 = vp[0], v1 = vp[1];
            #pragma unroll
            for (int e = 0; e < 8; ++e) Vts[(cseg + e) * LSTR + crow]     = v0[e];
            #pragma unroll
            for (int e = 0; e < 8; ++e) Vts[(cseg + 8 + e) * LSTR + crow] = v1[e];
        }
        if (kb + KC < Sq) {
            __builtin_prefetch(Kb + (size_t)(kb + KC + crow) * Dk + cseg, 0, 1);
            __builtin_prefetch(Vb + (size_t)(kb + KC + crow) * Dk + cseg, 0, 1);
        }
        async_stage_wait();
        __syncthreads();

        // ---- scores: S = Q (16x64) x K^T (64x64) -> 4 tiles of 16x16 ----
        v8f st[4] = {};
        #pragma unroll
        for (int t = 0; t < 4; ++t) {
            const bf16_t* r = &Ks[(t * 16 + ln) * LSTR];
            v16bf b0 = ld_frag(r + half * 8,      r + 16 + half * 8);       // k 0..31
            v16bf b1 = ld_frag(r + 32 + half * 8, r + 48 + half * 8);  // k 32..63
            st[t] = wmma_bf16(qa[0], b0, st[t]);
            st[t] = wmma_bf16(qa[1], b1, st[t]);
        }

        // ---- scale + mask ----
        #pragma unroll
        for (int t = 0; t < 4; ++t) {
            #pragma unroll
            for (int i = 0; i < 8; ++i) {
                int qrow = q0 + wid * 16 + i + half * 8;
                int kcol = kb + t * 16 + ln;
                int mv = mask[((size_t)b * Sq + qrow) * Sq + kcol];
                float sc = st[t][i] * 0.125f;              // 1/sqrt(64)
                st[t][i] = mv ? sc : -1.0e9f;
            }
        }

        // ---- online softmax (row stats replicated across each 16-lane half) ----
        float mnew[8], alpha[8], rs[8];
        #pragma unroll
        for (int i = 0; i < 8; ++i) {
            float v = fmaxf(fmaxf(st[0][i], st[1][i]), fmaxf(st[2][i], st[3][i]));
            v = fmaxf(v, __shfl_xor(v, 1, 32));
            v = fmaxf(v, __shfl_xor(v, 2, 32));
            v = fmaxf(v, __shfl_xor(v, 4, 32));
            v = fmaxf(v, __shfl_xor(v, 8, 32));
            mnew[i]  = fmaxf(m_run[i], v);
            alpha[i] = exp2f((m_run[i] - mnew[i]) * LOG2E);
            m_run[i] = mnew[i];
            rs[i] = 0.0f;
        }
        // ---- P = exp(S - m), write bf16 P to per-wave LDS, accumulate row sums ----
        #pragma unroll
        for (int t = 0; t < 4; ++t) {
            #pragma unroll
            for (int i = 0; i < 8; ++i) {
                float p = exp2f((st[t][i] - mnew[i]) * LOG2E);
                rs[i] += p;
                Pw[(i + half * 8) * LSTR + t * 16 + ln] = (bf16_t)p;
            }
        }
        #pragma unroll
        for (int i = 0; i < 8; ++i) {
            float r = rs[i];
            r += __shfl_xor(r, 1, 32);
            r += __shfl_xor(r, 2, 32);
            r += __shfl_xor(r, 4, 32);
            r += __shfl_xor(r, 8, 32);
            l_run[i] = l_run[i] * alpha[i] + r;
        }
        #pragma unroll
        for (int dt = 0; dt < 4; ++dt)
            #pragma unroll
            for (int i = 0; i < 8; ++i)
                o[dt][i] *= alpha[i];

        // ---- O += P (16x64) x V (64x64): reload P as A-fragments (same-wave LDS in-order) ----
        const bf16_t* pr = &Pw[ln * LSTR];
        v16bf pa[2];
        pa[0] = ld_frag(pr + half * 8,      pr + 16 + half * 8);
        pa[1] = ld_frag(pr + 32 + half * 8, pr + 48 + half * 8);
        #pragma unroll
        for (int dt = 0; dt < 4; ++dt) {
            const bf16_t* vr = &Vts[(dt * 16 + ln) * LSTR];
            v16bf b0 = ld_frag(vr + half * 8,      vr + 16 + half * 8);       // key 0..31
            v16bf b1 = ld_frag(vr + 32 + half * 8, vr + 48 + half * 8);  // key 32..63
            o[dt] = wmma_bf16(pa[0], b0, o[dt]);
            o[dt] = wmma_bf16(pa[1], b1, o[dt]);
        }
    }

    // ---- normalize and write attn output in [B,S,H*64] bf16 (row-major for final GEMM) ----
    float inv[8];
    #pragma unroll
    for (int i = 0; i < 8; ++i) inv[i] = 1.0f / fmaxf(l_run[i], 1e-30f);
    #pragma unroll
    for (int dt = 0; dt < 4; ++dt) {
        #pragma unroll
        for (int i = 0; i < 8; ++i) {
            int qrow = q0 + wid * 16 + i + half * 8;
            int col  = h * Dk + dt * 16 + ln;
            AO[((size_t)b * Sq + qrow) * Dm + col] = (bf16_t)(o[dt][i] * inv[i]);
        }
    }
}

// ---------------------------------------------------------------------------
// Host-side orchestration
// ---------------------------------------------------------------------------
extern "C" void kernel_launch(void* const* d_in, const int* in_sizes, int n_in,
                              void* d_out, int out_size, void* d_ws, size_t ws_size,
                              hipStream_t stream) {
    (void)in_sizes; (void)n_in; (void)out_size; (void)ws_size;
    const float* q    = (const float*)d_in[0];
    const float* k    = (const float*)d_in[1];
    const float* v    = (const float*)d_in[2];
    const int*   mask = (const int*)  d_in[3];
    const float* Wq   = (const float*)d_in[4];
    const float* Wk   = (const float*)d_in[5];
    const float* Wv   = (const float*)d_in[6];
    const float* Wo   = (const float*)d_in[7];

    constexpr size_t WE = (size_t)Dm * Dm;       // 1M elements per weight
    constexpr size_t QE = (size_t)Mrows * Dm;    // 8M elements per activation
    bf16_t* base = (bf16_t*)d_ws;
    bf16_t* wq_b = base;
    bf16_t* wk_b = base + 1 * WE;
    bf16_t* wv_b = base + 2 * WE;
    bf16_t* wo_b = base + 3 * WE;
    bf16_t* Qb   = base + 4 * WE;
    bf16_t* Kb   = Qb + QE;
    bf16_t* Vb   = Kb + QE;
    bf16_t* AO   = Vb + QE;

    // 1) pack weights to bf16
    const int n4 = (int)(WE / 4);
    dim3 pgrd((n4 + 255) / 256), pblk(256);
    pack_bf16_kernel<<<pgrd, pblk, 0, stream>>>(Wq, wq_b, n4);
    pack_bf16_kernel<<<pgrd, pblk, 0, stream>>>(Wk, wk_b, n4);
    pack_bf16_kernel<<<pgrd, pblk, 0, stream>>>(Wv, wv_b, n4);
    pack_bf16_kernel<<<pgrd, pblk, 0, stream>>>(Wo, wo_b, n4);

    // 2) Q/K/V projections: [8192,1024] @ W^T -> bf16 [B,H,S,64]
    dim3 ggrd(Dm / 128, Mrows / 128), gblk(256);
    gemm_xw_kernel<true, true><<<ggrd, gblk, 0, stream>>>(q, wq_b, Qb);
    gemm_xw_kernel<true, true><<<ggrd, gblk, 0, stream>>>(k, wk_b, Kb);
    gemm_xw_kernel<true, true><<<ggrd, gblk, 0, stream>>>(v, wv_b, Vb);

    // 3) attention (flash, online softmax)
    dim3 agrd(Sq / 128, Hh, Bsz), ablk(256);
    mha_flash_kernel<<<agrd, ablk, 0, stream>>>(Qb, Kb, Vb, mask, AO);

    // 4) output projection: attn(bf16) @ Wo^T -> f32 d_out
    gemm_xw_kernel<false, false><<<ggrd, gblk, 0, stream>>>(AO, wo_b, (float*)d_out);
}
